// CorefMentionRankingHead_22213570855160
// MI455X (gfx1250) — matrix operand order repaired and checked by reading, not compiled, not run on hardware
//
#include <hip/hip_runtime.h>
#include <hip/hip_bf16.h>
#include <stdint.h>

#define N_MENT 2048
#define HID    1024
#define FEAT   20
#define MAXA   50
#define OUTC   51

typedef __attribute__((ext_vector_type(16))) __bf16    v16bf;
typedef __attribute__((ext_vector_type(8)))  float     v8f;
typedef __attribute__((ext_vector_type(4)))  uint32_t  u32x4;

union Frag {
    u32x4    q[2];
    uint32_t u[8];
    v16bf    v;
};

// fp32 -> bf16 round-to-nearest-even, result in low 16 bits
__device__ __forceinline__ uint32_t f2bf(float f) {
    uint32_t u = __float_as_uint(f);
    return (u + 0x7FFFu + ((u >> 16) & 1u)) >> 16;
}

// packed bf16x2 multiply (two bf16 in a dword)
__device__ __forceinline__ uint32_t bfmul2(uint32_t a, uint32_t b) {
    float alo = __uint_as_float(a << 16);
    float ahi = __uint_as_float(a & 0xFFFF0000u);
    float blo = __uint_as_float(b << 16);
    float bhi = __uint_as_float(b & 0xFFFF0000u);
    return (f2bf(ahi * bhi) << 16) | (f2bf(alo * blo) & 0xFFFFu);
}

__device__ __forceinline__ v8f wmma_bf16(const Frag& a, const Frag& b, v8f c) {
    return __builtin_amdgcn_wmma_f32_16x16x32_bf16(
        false, a.v, false, b.v, (short)0, c, false, false);
}

// ---------------------------------------------------------------- conversions

__global__ void k_cvt_bf16(const float* __restrict__ src,
                           uint16_t* __restrict__ dst, int n) {
    int t = blockIdx.x * blockDim.x + threadIdx.x;
    if (t < n) dst[t] = (uint16_t)f2bf(src[t]);
}

// Wt[b][h][d] = bf16(W1[(b*HID + d)*HID + h])  for b in {0,1,2} (cur, ant, mul)
__global__ void k_w1_transpose(const float* __restrict__ W1,
                               uint16_t* __restrict__ Wt) {
    int t   = blockIdx.x * blockDim.x + threadIdx.x;
    int b   = t >> 20;
    int rem = t & ((1 << 20) - 1);
    int h   = rem >> 10;
    int d   = rem & (HID - 1);
    Wt[(size_t)b * HID * HID + (size_t)h * HID + d] =
        (uint16_t)f2bf(W1[(size_t)(b * HID + d) * HID + h]);
}

// Pdf[z][h] = b1[h] + sum_f dist_emb[z,f] * W1[3H+f, h]
__global__ void k_pdf(const float* __restrict__ W1, const float* __restrict__ b1,
                      const float* __restrict__ dist_emb, float* __restrict__ Pdf) {
    int t = blockIdx.x * blockDim.x + threadIdx.x;
    if (t >= 10 * HID) return;
    int z = t >> 10, h = t & (HID - 1);
    float acc = b1[h];
#pragma unroll
    for (int f = 0; f < FEAT; ++f)
        acc += dist_emb[z * FEAT + f] * W1[(size_t)(3 * HID + f) * HID + h];
    Pdf[t] = acc;
}

// ------------------------------------------------- precompute GEMM (Pcur/Pant)
// P[n,h] = sum_d Abf[n,d] * Wt[h,d]   (Abf row-major bf16, Wt col-major bf16)
// Double-buffered: loads for chunk k+1 overlap WMMAs of chunk k.
__global__ void __launch_bounds__(256)
k_gemm_pre(const uint16_t* __restrict__ Abf, const uint16_t* __restrict__ Wt,
           float* __restrict__ P) {
    const int L     = threadIdx.x & 31;
    const int w     = threadIdx.x >> 5;
    const int rt    = blockIdx.x;
    const int m     = L & 15;
    const int khalf = (L >> 4) * 8;
    const int kbB   = (L >> 4) * 8;
    const int mhi   = (L >> 4) * 8;

    const uint16_t* aRow = Abf + (size_t)(rt * 16 + m) * HID;

#pragma unroll 1
    for (int g = 0; g < 2; ++g) {
        v8f acc[4];
#pragma unroll
        for (int c = 0; c < 4; ++c) acc[c] = {};

        const uint16_t* bp[4];
#pragma unroll
        for (int c = 0; c < 4; ++c)
            bp[c] = Wt + (size_t)((g * 32 + w * 4 + c) * 16 + m) * HID + kbB;

        Frag A0, A1, B0[4], B1[4];
        // prologue: chunk 0 into buffer 0
        A0.q[0] = *(const u32x4*)(aRow + khalf);
        A0.q[1] = *(const u32x4*)(aRow + khalf + 16);
#pragma unroll
        for (int c = 0; c < 4; ++c) {
            B0[c].q[0] = *(const u32x4*)(bp[c]);
            B0[c].q[1] = *(const u32x4*)(bp[c] + 16);
        }

#pragma unroll 1
        for (int kk = 0; kk < 16; ++kk) {
            const int kA = kk * 64, kB = kA + 32, kC = kA + 64;
            // load chunk kB into buffer 1
            A1.q[0] = *(const u32x4*)(aRow + kB + khalf);
            A1.q[1] = *(const u32x4*)(aRow + kB + khalf + 16);
#pragma unroll
            for (int c = 0; c < 4; ++c) {
                B1[c].q[0] = *(const u32x4*)(bp[c] + kB);
                B1[c].q[1] = *(const u32x4*)(bp[c] + kB + 16);
            }
            // compute chunk kA from buffer 0
#pragma unroll
            for (int c = 0; c < 4; ++c) acc[c] = wmma_bf16(A0, B0[c], acc[c]);
            // load chunk kC into buffer 0
            if (kk < 15) {
                A0.q[0] = *(const u32x4*)(aRow + kC + khalf);
                A0.q[1] = *(const u32x4*)(aRow + kC + khalf + 16);
#pragma unroll
                for (int c = 0; c < 4; ++c) {
                    B0[c].q[0] = *(const u32x4*)(bp[c] + kC);
                    B0[c].q[1] = *(const u32x4*)(bp[c] + kC + 16);
                }
            }
            // compute chunk kB from buffer 1
#pragma unroll
            for (int c = 0; c < 4; ++c) acc[c] = wmma_bf16(A1, B1[c], acc[c]);
        }

#pragma unroll
        for (int c = 0; c < 4; ++c) {
            int col = (g * 32 + w * 4 + c) * 16 + (L & 15);
#pragma unroll
            for (int v = 0; v < 8; ++v)
                P[(size_t)(rt * 16 + v + mhi) * HID + col] = acc[c][v];
        }
    }
}

// ------------------------------------------------------------- main scorer
// One WG per (mention i, 16 padded slots). A tile (cur*ant, bf16) is built
// once into LDS; waves read A fragments via ds_load_b128. B stream is
// double-buffered from global (L2-resident W1_mul).
__global__ void __launch_bounds__(256)
k_score(const uint32_t* __restrict__ Ebf2,     // Ebf viewed as dwords [N][512]
        const uint16_t* __restrict__ WtMul,
        const float* __restrict__ Pcur, const float* __restrict__ Pant,
        const float* __restrict__ Pdf, const float* __restrict__ W2,
        const float* __restrict__ b2, float* __restrict__ out) {
    __shared__ uint32_t ldsM[16 * 512];        // 16 rows x 1024 bf16 = 32 KB
    __shared__ float    s_row[16];

    const int tid = threadIdx.x;
    const int L   = tid & 31;
    const int w   = tid >> 5;
    const int i   = blockIdx.x >> 2;
    const int a0  = (blockIdx.x & 3) * 16;
    const int off = (i > MAXA) ? (i - MAXA) : 0;

    if (tid < 16) s_row[tid] = 0.0f;

    // cooperative build of the product tile: ldsM[r][p] = bf16x2(cur*ant)
#pragma unroll 4
    for (int idx = tid; idx < 16 * 512; idx += 256) {
        int r  = idx >> 9;
        int p  = idx & 511;
        int J  = off + a0 + r;
        int Jc = (J < N_MENT - 1) ? J : (N_MENT - 1);
        ldsM[idx] = bfmul2(Ebf2[(size_t)i * 512 + p], Ebf2[(size_t)Jc * 512 + p]);
    }
    __syncthreads();

    const int m     = L & 15;
    const int khalf = (L >> 4) * 8;
    const int kbB   = (L >> 4) * 8;
    const int mhi   = (L >> 4) * 8;

    const uint16_t* ldsRow = (const uint16_t*)ldsM + m * HID;

    float sp[8];
#pragma unroll
    for (int v = 0; v < 8; ++v) sp[v] = 0.0f;

#pragma unroll 1
    for (int g = 0; g < 2; ++g) {
        v8f acc[4];
#pragma unroll
        for (int c = 0; c < 4; ++c) acc[c] = {};

        const uint16_t* bp[4];
#pragma unroll
        for (int c = 0; c < 4; ++c)
            bp[c] = WtMul + (size_t)((g * 32 + w * 4 + c) * 16 + m) * HID + kbB;

        Frag A0, A1, B0[4], B1[4];
        A0.q[0] = *(const u32x4*)(ldsRow + khalf);
        A0.q[1] = *(const u32x4*)(ldsRow + khalf + 16);
#pragma unroll
        for (int c = 0; c < 4; ++c) {
            B0[c].q[0] = *(const u32x4*)(bp[c]);
            B0[c].q[1] = *(const u32x4*)(bp[c] + 16);
        }

#pragma unroll 1
        for (int kk = 0; kk < 16; ++kk) {
            const int kA = kk * 64, kB = kA + 32, kC = kA + 64;
            A1.q[0] = *(const u32x4*)(ldsRow + kB + khalf);
            A1.q[1] = *(const u32x4*)(ldsRow + kB + khalf + 16);
#pragma unroll
            for (int c = 0; c < 4; ++c) {
                B1[c].q[0] = *(const u32x4*)(bp[c] + kB);
                B1[c].q[1] = *(const u32x4*)(bp[c] + kB + 16);
            }
#pragma unroll
            for (int c = 0; c < 4; ++c) acc[c] = wmma_bf16(A0, B0[c], acc[c]);
            if (kk < 15) {
                A0.q[0] = *(const u32x4*)(ldsRow + kC + khalf);
                A0.q[1] = *(const u32x4*)(ldsRow + kC + khalf + 16);
#pragma unroll
                for (int c = 0; c < 4; ++c) {
                    B0[c].q[0] = *(const u32x4*)(bp[c] + kC);
                    B0[c].q[1] = *(const u32x4*)(bp[c] + kC + 16);
                }
            }
#pragma unroll
            for (int c = 0; c < 4; ++c) acc[c] = wmma_bf16(A1, B1[c], acc[c]);
        }

        // epilogue for this column group
#pragma unroll
        for (int c = 0; c < 4; ++c) {
            int col   = (g * 32 + w * 4 + c) * 16 + (L & 15);
            float pc  = Pcur[(size_t)i * HID + col];
            float w2c = W2[col];
#pragma unroll
            for (int v = 0; v < 8; ++v) {
                int mm  = v + mhi;
                int Jv  = off + a0 + mm;
                int Jvc = (Jv < N_MENT - 1) ? Jv : (N_MENT - 1);
                int dv  = i - Jv;
                dv = dv < 0 ? 0 : (dv > 9 ? 9 : dv);
                float h = acc[c][v] + pc + Pant[(size_t)Jvc * HID + col]
                        + Pdf[dv * HID + col];
                h = h > 0.0f ? h : 0.0f;
                sp[v] += h * w2c;
            }
        }
    }

    // reduce across the 16 lanes sharing each row, then across waves via LDS
#pragma unroll
    for (int v = 0; v < 8; ++v) {
        float s = sp[v];
        s += __shfl_xor(s, 1, 16);
        s += __shfl_xor(s, 2, 16);
        s += __shfl_xor(s, 4, 16);
        s += __shfl_xor(s, 8, 16);
        if ((L & 15) == 0) atomicAdd(&s_row[v + mhi], s);
    }
    __syncthreads();

    if (tid < 16) {
        int slot = a0 + tid;
        if (slot < MAXA) {
            int nvalid = i < MAXA ? i : MAXA;
            float val = (slot < nvalid) ? (s_row[tid] + b2[0]) : 0.0f;
            out[(size_t)i * OUTC + 1 + slot] = val;
        }
    } else if (tid == 16 && a0 == 0) {
        out[(size_t)i * OUTC] = 0.0f;          // dummy antecedent
    }
}

// ------------------------------------------------------------------ launcher

extern "C" void kernel_launch(void* const* d_in, const int* in_sizes, int n_in,
                              void* d_out, int out_size, void* d_ws, size_t ws_size,
                              hipStream_t stream) {
    const float* E        = (const float*)d_in[0];
    const float* W1       = (const float*)d_in[3];
    const float* b1       = (const float*)d_in[4];
    const float* W2       = (const float*)d_in[5];
    const float* b2       = (const float*)d_in[6];
    const float* dist_emb = (const float*)d_in[7];
    float* out            = (float*)d_out;

    uint8_t* ws = (uint8_t*)d_ws;
    uint16_t* Ebf = (uint16_t*)(ws);                              // 4 MB
    uint16_t* Wt  = (uint16_t*)(ws + (size_t)4  * 1024 * 1024);   // 6 MB (3 blocks)
    float*   Pcur = (float*)  (ws + (size_t)10 * 1024 * 1024);    // 8 MB
    float*   Pant = (float*)  (ws + (size_t)18 * 1024 * 1024);    // 8 MB
    float*   Pdf  = (float*)  (ws + (size_t)26 * 1024 * 1024);    // 40 KB

    k_cvt_bf16<<<(N_MENT * HID) / 256, 256, 0, stream>>>(E, Ebf, N_MENT * HID);
    k_w1_transpose<<<(3 * HID * HID) / 256, 256, 0, stream>>>(W1, Wt);
    k_pdf<<<(10 * HID + 255) / 256, 256, 0, stream>>>(W1, b1, dist_emb, Pdf);

    k_gemm_pre<<<N_MENT / 16, 256, 0, stream>>>(Ebf, Wt,                     Pcur);
    k_gemm_pre<<<N_MENT / 16, 256, 0, stream>>>(Ebf, Wt + (size_t)HID * HID, Pant);

    k_score<<<N_MENT * 4, 256, 0, stream>>>((const uint32_t*)Ebf,
                                            Wt + (size_t)2 * HID * HID,
                                            Pcur, Pant, Pdf, W2, b2, out);
}
